// LSM_75411035783552
// MI455X (gfx1250) — matrix operand order repaired and checked by reading, not compiled, not run on hardware
//
#include <hip/hip_runtime.h>

// ---------------------------------------------------------------------------
// LSM step, t=0, zero initial state  ==>  spk = (x0 @ Win^T + b1 + brec) > 1
//   x0 : [256, 8192]  (x[:,0] flattened; x row stride is T*C*H*W = 65536 f32)
//   Win: [4096, 8192] ; b1, brec: [4096] ; Wrec is mathematically unused.
// Single fused f16-WMMA GEMM (f32 accumulate) with bias+threshold epilogue.
// ---------------------------------------------------------------------------

typedef __attribute__((ext_vector_type(16))) _Float16 v16h;
typedef __attribute__((ext_vector_type(8)))  _Float16 v8h;
typedef __attribute__((ext_vector_type(4)))  _Float16 v4h;
typedef __attribute__((ext_vector_type(8)))  float    v8f;

#define N_RES        4096
#define IN_SZ        8192
#define BATCH        256
#define X_ROW_STRIDE 65536     // 8*2*64*64 floats between batch rows of x
#define KTILE        32
#define KSTEPS       (IN_SZ / KTILE)   // 256
#define LDH          40        // LDS row stride in halfs (32 data + 8 pad; 80B, 16B-aligned)
#define MT           128       // M rows per workgroup
#define NT           128       // N cols per workgroup

__global__ __launch_bounds__(256)
void lsm_spk_gemm(const float* __restrict__ x,
                  const float* __restrict__ Win,
                  const float* __restrict__ b1,
                  const float* __restrict__ brec,
                  float* __restrict__ out)
{
    __shared__ _Float16 As[2][MT * LDH];   // 128 rows x 32 k (f16)
    __shared__ _Float16 Bs[2][NT * LDH];   // 128 Win-rows x 32 k (f16)

    const int tid   = threadIdx.x;
    const int lane  = tid & 31;
    const int wave  = tid >> 5;      // 0..7
    const int mwave = wave & 3;      // 4 waves along M, 32 rows each
    const int nwave = wave >> 2;     // 2 waves along N, 64 cols each
    const int lrow  = lane & 15;
    const int lhalf = lane >> 4;

    const int mtb = blockIdx.y * MT;
    const int ntb = blockIdx.x * NT;

    // ---- staging: 4 float4 per tile per thread (128x32 f32 = 1024 float4 / 256 thr)
    float4 ra[4], rb[4];

    auto loadRegs = [&](int kt) {
        const int kb = kt * KTILE;
        #pragma unroll
        for (int i = 0; i < 4; ++i) {
            const int id  = tid + i * 256;
            const int row = id >> 3;
            const int kq  = id & 7;
            ra[i] = *reinterpret_cast<const float4*>(
                        x + (size_t)(mtb + row) * X_ROW_STRIDE + kb + kq * 4);
            rb[i] = *reinterpret_cast<const float4*>(
                        Win + (size_t)(ntb + row) * IN_SZ + kb + kq * 4);
        }
    };

    auto storeLDS = [&](int buf) {
        #pragma unroll
        for (int i = 0; i < 4; ++i) {
            const int id  = tid + i * 256;
            const int row = id >> 3;
            const int kq  = id & 7;
            v4h ha = { (_Float16)ra[i].x, (_Float16)ra[i].y,
                       (_Float16)ra[i].z, (_Float16)ra[i].w };
            v4h hb = { (_Float16)rb[i].x, (_Float16)rb[i].y,
                       (_Float16)rb[i].z, (_Float16)rb[i].w };
            *reinterpret_cast<v4h*>(&As[buf][row * LDH + kq * 4]) = ha;
            *reinterpret_cast<v4h*>(&Bs[buf][row * LDH + kq * 4]) = hb;
        }
    };

    v8f acc[2][4];
    #pragma unroll
    for (int mi = 0; mi < 2; ++mi)
        #pragma unroll
        for (int ni = 0; ni < 4; ++ni)
            #pragma unroll
            for (int r = 0; r < 8; ++r)
                acc[mi][ni][r] = 0.0f;

    auto compute = [&](int buf) {
        const _Float16* aB = &As[buf][0];
        const _Float16* bB = &Bs[buf][0];
        v16h af[2], bf[4];
        // A fragment (16x32 f16): lane holds row lane%16, K chunks {lhalf*8, +16}
        #pragma unroll
        for (int mi = 0; mi < 2; ++mi) {
            const _Float16* p = aB + (mwave * 32 + mi * 16 + lrow) * LDH + lhalf * 8;
            v8h lo = *reinterpret_cast<const v8h*>(p);
            v8h hi = *reinterpret_cast<const v8h*>(p + 16);
            af[mi] = __builtin_shufflevector(lo, hi,
                        0,1,2,3,4,5,6,7,8,9,10,11,12,13,14,15);
        }
        // B fragment (32x16 f16 = Win^T tile): lane holds Win row lane%16,
        // K chunks {lhalf*8, +16}
        #pragma unroll
        for (int ni = 0; ni < 4; ++ni) {
            const _Float16* p = bB + (nwave * 64 + ni * 16 + lrow) * LDH + lhalf * 8;
            v8h lo = *reinterpret_cast<const v8h*>(p);
            v8h hi = *reinterpret_cast<const v8h*>(p + 16);
            bf[ni] = __builtin_shufflevector(lo, hi,
                        0,1,2,3,4,5,6,7,8,9,10,11,12,13,14,15);
        }
        #pragma unroll
        for (int mi = 0; mi < 2; ++mi)
            #pragma unroll
            for (int ni = 0; ni < 4; ++ni)
                acc[mi][ni] = __builtin_amdgcn_wmma_f32_16x16x32_f16(
                    /*neg_a=*/false, af[mi],
                    /*neg_b=*/false, bf[ni],
                    /*c_mod=*/(short)0, acc[mi][ni],
                    /*reuse_a=*/false, /*reuse_b=*/false);
    };

    // ---- software pipeline: global load (kt+1) || wmma(kt) ; cvt+LDS store ; barrier
    loadRegs(0);
    storeLDS(0);
    __syncthreads();

    int buf = 0;
    #pragma unroll 1
    for (int kt = 0; kt < KSTEPS; ++kt) {
        if (kt + 4 < KSTEPS) {  // gfx1250 global_prefetch_b8 a few tiles ahead
            const int kb = (kt + 4) * KTILE;
            __builtin_prefetch(x   + (size_t)(mtb + (tid >> 3)) * X_ROW_STRIDE + kb, 0, 0);
            __builtin_prefetch(Win + (size_t)(ntb + (tid >> 3)) * IN_SZ        + kb, 0, 0);
        }
        if (kt + 1 < KSTEPS) loadRegs(kt + 1);   // in flight during WMMA
        compute(buf);
        if (kt + 1 < KSTEPS) storeLDS(buf ^ 1);
        __syncthreads();
        buf ^= 1;
    }

    // ---- epilogue: mem = acc + b1 + brec ; spk = mem > 1.0
    // C layout: VGPR r, lane l -> M = r + 8*(l/16), N = l%16
    #pragma unroll
    for (int ni = 0; ni < 4; ++ni) {
        const int col  = ntb + nwave * 64 + ni * 16 + lrow;
        const float bias = b1[col] + brec[col];
        #pragma unroll
        for (int mi = 0; mi < 2; ++mi) {
            #pragma unroll
            for (int r = 0; r < 8; ++r) {
                const int row = mtb + mwave * 32 + mi * 16 + lhalf * 8 + r;
                const float mem = acc[mi][ni][r] + bias;
                out[(size_t)row * N_RES + col] = (mem > 1.0f) ? 1.0f : 0.0f;
            }
        }
    }
}

extern "C" void kernel_launch(void* const* d_in, const int* in_sizes, int n_in,
                              void* d_out, int out_size, void* d_ws, size_t ws_size,
                              hipStream_t stream) {
    // setup_inputs order: x, Win, b1, Wrec, brec
    const float* x    = (const float*)d_in[0];
    const float* Win  = (const float*)d_in[1];
    const float* b1   = (const float*)d_in[2];
    // d_in[3] = Wrec: unused (spk0 == 0  =>  spk0 @ Wrec.T == 0)
    const float* brec = (const float*)d_in[4];
    float* out = (float*)d_out;

    dim3 grid(N_RES / NT, BATCH / MT);   // (32, 2)
    lsm_spk_gemm<<<grid, dim3(256), 0, stream>>>(x, Win, b1, brec, out);
}
